// Model_78821239816770
// MI455X (gfx1250) — compile-verified, matrix operand order checked by reading, not run on hardware
//
#include <hip/hip_runtime.h>
#include <hip/hip_bf16.h>
#include <math.h>

// ---------------------------------------------------------------------------
// FEDformer forward for MI455X (gfx1250). All big matmuls (projections, FFN,
// and the 64-mode DFT/iDFT recast as GEMMs) go through v_wmma_f32_16x16x32_bf16
// with f32 accumulation; bandwidth-bound pieces are plain streaming kernels.
// GEMM: reg->LDS double-buffered pipeline, branch-free templated epilogue,
// L2 prefetch two tiles ahead.
// ---------------------------------------------------------------------------

typedef __bf16 bf16_t;
typedef bf16_t v16bf __attribute__((ext_vector_type(16)));
typedef float  v8f  __attribute__((ext_vector_type(8)));

#define BBATCH 16
#define LLEN   2048
#define DMODEL 512
#define NHEADS 8
#define EHEAD  64
#define NMODES 64
#define DFF    2048
#define CCH    21
#define NSEQ   (BBATCH*NHEADS*EHEAD)   /* 8192 */

__device__ __forceinline__ bf16_t f2bf(float f) {
  unsigned u = __builtin_bit_cast(unsigned, f);
  u += 0x7fffu + ((u >> 16) & 1u);               // round-to-nearest-even
  unsigned short s = (unsigned short)(u >> 16);
  return __builtin_bit_cast(bf16_t, s);
}

// ---------------------------------------------------------------------------
// WMMA GEMM:  C[m,n] (+)= sum_k A[m,k] * W[n,k]  (+ bias[n])
// A: [M,K] f32 row-major, W: [N,K] f32 row-major, C: [M,N] f32 row-major.
// Block tile 64x64, BLK_K=32, 4 waves (128 thr), each wave 2x2 WMMA tiles.
// Double-buffered LDS staging, one barrier per K-step.
// M,N multiples of 64; K multiple of 32 (true for all call sites).
// ---------------------------------------------------------------------------
template <bool ADD, bool HASBIAS>
__global__ __launch_bounds__(128)
void k_gemm_bf16(const float* __restrict__ A, const float* __restrict__ W,
                 const float* __restrict__ bias, float* __restrict__ C,
                 int M, int N, int K)
{
  __shared__ bf16_t As[2][64][40];   // pad to 40 (80B rows, 16B aligned)
  __shared__ bf16_t Bs[2][64][40];

  const int tid  = threadIdx.x;
  const int wave = tid >> 5;
  const int lane = tid & 31;
  const int wr   = wave >> 1;      // wave row (0..1) -> 32 rows
  const int wc   = wave & 1;       // wave col (0..1) -> 32 cols
  const int bm   = blockIdx.y * 64;
  const int bn   = blockIdx.x * 64;

  v8f acc[2][2] = {};              // zero-initialized accumulators

  const int ar = tid >> 1;         // 0..63 row within tile
  const int ac = (tid & 1) * 16;   // 0 or 16
  const int h  = lane >> 4;
  const int lm = lane & 15;

  const float* aRow = A + (size_t)(bm + ar) * K + ac;
  const float* wRow = W + (size_t)(bn + ar) * K + ac;

  // prologue: tile 0 into registers (4x b128 each)
  float4 ra[4], rw[4];
  {
    const float4* ap = (const float4*)aRow;
    const float4* wp = (const float4*)wRow;
#pragma unroll
    for (int j = 0; j < 4; ++j) { ra[j] = ap[j]; rw[j] = wp[j]; }
  }

  const int T = K >> 5;
  for (int i = 0; i < T; ++i) {
    const int buf = i & 1;
    // stage registers -> LDS as bf16
    const float* raf = reinterpret_cast<const float*>(ra);
    const float* rwf = reinterpret_cast<const float*>(rw);
#pragma unroll
    for (int j = 0; j < 16; ++j) {
      As[buf][ar][ac + j] = f2bf(raf[j]);
      Bs[buf][ar][ac + j] = f2bf(rwf[j]);
    }
    __syncthreads();

    // prefetch next tile into registers (overlaps WMMA below)
    if (i + 1 < T) {
      const float4* ap = (const float4*)(aRow + (size_t)(i + 1) * 32);
      const float4* wp = (const float4*)(wRow + (size_t)(i + 1) * 32);
#pragma unroll
      for (int j = 0; j < 4; ++j) { ra[j] = ap[j]; rw[j] = wp[j]; }
      if (i + 2 < T) {
        __builtin_prefetch(aRow + (size_t)(i + 2) * 32, 0, 0);  // global_prefetch_b8
        __builtin_prefetch(wRow + (size_t)(i + 2) * 32, 0, 0);
      }
    }

    // Fragment build per ISA 7.12.2 (16-bit A 16x32 / B 32x16 layouts):
    // lane half h, element e -> k = (e<8?0:16) + h*8 + (e&7)  (two b128 runs)
    v16bf afrag[2], bfrag[2];
#pragma unroll
    for (int t = 0; t < 2; ++t) {
      const int rowA = wr * 32 + t * 16 + lm;
      const int rowB = wc * 32 + t * 16 + lm;
#pragma unroll
      for (int e = 0; e < 16; ++e) {
        const int kk = ((e < 8) ? 0 : 16) + h * 8 + (e & 7);
        afrag[t][e] = As[buf][rowA][kk];
        bfrag[t][e] = Bs[buf][rowB][kk];
      }
    }
#pragma unroll
    for (int ii = 0; ii < 2; ++ii)
#pragma unroll
      for (int j = 0; j < 2; ++j)
        acc[ii][j] = __builtin_amdgcn_wmma_f32_16x16x32_bf16(
            false, afrag[ii], false, bfrag[j], (short)0, acc[ii][j], false, false);
    // single barrier per iteration is sufficient: buffer b written at iter i
    // was last read at iter i-2; a barrier separates them.
  }

  // Epilogue (branch-free): lane: n = lane%16, vgpr r: m = r + 8*(lane/16)
#pragma unroll
  for (int i = 0; i < 2; ++i)
#pragma unroll
    for (int j = 0; j < 2; ++j) {
      const int n = bn + wc * 32 + j * 16 + lm;
      float bv = 0.0f;
      if (HASBIAS) bv = bias[n];
#pragma unroll
      for (int r = 0; r < 8; ++r) {
        const int m = bm + wr * 32 + i * 16 + r + h * 8;
        const size_t idx = (size_t)m * N + n;
        float v = acc[i][j][r] + bv;
        if (ADD) v += C[idx];
        C[idx] = v;
      }
    }
}

// ---------------------------------------------------------------------------
// Per-batch transpose [L,512] -> [512,L] (for DFT along L).
// ---------------------------------------------------------------------------
__global__ void k_transpose_bl(const float* __restrict__ in, float* __restrict__ out)
{
  __shared__ float t[32][33];
  const int b  = blockIdx.z;
  const int l0 = blockIdx.x * 32;
  const int d0 = blockIdx.y * 32;
  const int lx = threadIdx.x, ly = threadIdx.y;
  const float* ip = in  + (size_t)b * LLEN * DMODEL;
  float*       op = out + (size_t)b * DMODEL * LLEN;
  for (int i = ly; i < 32; i += 8)
    t[i][lx] = ip[(size_t)(l0 + i) * DMODEL + d0 + lx];
  __syncthreads();
  for (int i = ly; i < 32; i += 8)
    op[(size_t)(d0 + i) * LLEN + l0 + lx] = t[lx][i];
}

// DFT / iDFT bases (phase reduced mod L for accuracy).
// fb[2m][l]=cos, fb[2m+1][l]=-sin ; ib[l][2m]=c_m cos, ib[l][2m+1]=-c_m sin
__global__ void k_gen_basis(float* __restrict__ fb, float* __restrict__ ib)
{
  int idx = blockIdx.x * blockDim.x + threadIdx.x;
  if (idx >= NMODES * LLEN) return;
  const int m = idx / LLEN, l = idx % LLEN;
  const int ph = (m * l) & (LLEN - 1);
  const float th = (float)ph * (6.283185307179586f / (float)LLEN);
  const float c = cosf(th), s = sinf(th);
  fb[(size_t)(2 * m) * LLEN + l]     = c;
  fb[(size_t)(2 * m + 1) * LLEN + l] = -s;
  const float cm = (m == 0) ? (1.0f / LLEN) : (2.0f / LLEN);
  ib[(size_t)l * (2 * NMODES) + 2 * m]     = cm * c;
  ib[(size_t)l * (2 * NMODES) + 2 * m + 1] = -cm * s;
}

// out[b,h,o,m] = scale * sum_i in[b,h,i,m] * (wre+j*wim)[h,i,o,m]
// coeff layout: [s = b*512 + h*64 + e][2m(+1)] interleaved re/im.
__global__ void k_mode_mix(const float* __restrict__ cin, const float* __restrict__ wre,
                           const float* __restrict__ wim, float* __restrict__ cout,
                           float scale)
{
  int idx = blockIdx.x * blockDim.x + threadIdx.x;
  if (idx >= BBATCH * NHEADS * EHEAD * NMODES) return;
  const int m = idx & 63, o = (idx >> 6) & 63, h = (idx >> 12) & 7, b = idx >> 15;
  const float* base = cin + (size_t)(b * 512 + h * 64) * 128;
  float ar = 0.f, ai = 0.f;
  for (int i = 0; i < 64; ++i) {
    const float xr = base[(size_t)i * 128 + 2 * m];
    const float xi = base[(size_t)i * 128 + 2 * m + 1];
    const size_t wi = (((size_t)(h * 64 + i)) * 64 + o) * 64 + m;
    const float wr = wre[wi], wm = wim[wi];
    ar += xr * wr - xi * wm;
    ai += xr * wm + xi * wr;
  }
  const size_t so = (size_t)(b * 512 + h * 64 + o) * 128;
  cout[so + 2 * m]     = ar * scale;
  cout[so + 2 * m + 1] = ai * scale;
}

// xqk[b,h,x,y] = ctanh( sum_e qf[b,h,e,x] * kf[b,h,e,y] )
__global__ void k_xqk_tanh(const float* __restrict__ cq, const float* __restrict__ ck,
                           float* __restrict__ xqk)
{
  int idx = blockIdx.x * blockDim.x + threadIdx.x;
  if (idx >= BBATCH * NHEADS * 64 * 64) return;
  const int y = idx & 63, x = (idx >> 6) & 63, h = (idx >> 12) & 7, b = idx >> 15;
  const float* qb = cq + (size_t)(b * 512 + h * 64) * 128;
  const float* kb = ck + (size_t)(b * 512 + h * 64) * 128;
  float ar = 0.f, ai = 0.f;
  for (int e = 0; e < 64; ++e) {
    const float qr = qb[(size_t)e * 128 + 2 * x], qi = qb[(size_t)e * 128 + 2 * x + 1];
    const float kr = kb[(size_t)e * 128 + 2 * y], ki = kb[(size_t)e * 128 + 2 * y + 1];
    ar += qr * kr - qi * ki;
    ai += qr * ki + qi * kr;
  }
  float re, im;
  if (fabsf(ar) > 20.f) { re = (ar > 0.f) ? 1.f : -1.f; im = 0.f; }
  else {
    const float den = coshf(2.f * ar) + cosf(2.f * ai);
    re = sinhf(2.f * ar) / den;
    im = sinf(2.f * ai) / den;
  }
  const size_t o = ((size_t)(b * 8 + h) * 4096 + x * 64 + y) * 2;
  xqk[o] = re; xqk[o + 1] = im;
}

// xqkv[b,h,e,x] = sum_y xqk[b,h,x,y] * kf[b,h,e,y]
__global__ void k_xqkv(const float* __restrict__ xqk, const float* __restrict__ ck,
                       float* __restrict__ cout)
{
  int idx = blockIdx.x * blockDim.x + threadIdx.x;
  if (idx >= BBATCH * NHEADS * EHEAD * 64) return;
  const int x = idx & 63, e = (idx >> 6) & 63, h = (idx >> 12) & 7, b = idx >> 15;
  const float* kb = ck + (size_t)(b * 512 + h * 64 + e) * 128;
  const float* qk = xqk + ((size_t)(b * 8 + h) * 4096 + x * 64) * 2;
  float ar = 0.f, ai = 0.f;
  for (int y = 0; y < 64; ++y) {
    const float tr = qk[2 * y], ti = qk[2 * y + 1];
    const float kr = kb[2 * y], ki = kb[2 * y + 1];
    ar += tr * kr - ti * ki;
    ai += tr * ki + ti * kr;
  }
  const size_t so = (size_t)(b * 512 + h * 64 + e) * 128;
  cout[so + 2 * x] = ar; cout[so + 2 * x + 1] = ai;
}

// circular conv1d, K=3: out[b,l,o] (+)= sum_{k,c} in[b,(l+k-1)%L,c]*w[o,c,k]
__global__ void k_conv3(const float* __restrict__ in, const float* __restrict__ w,
                        float* __restrict__ out, int Cin, int Cout, int addto)
{
  int idx = blockIdx.x * blockDim.x + threadIdx.x;
  const int total = BBATCH * LLEN * Cout;
  if (idx >= total) return;
  const int o = idx % Cout;
  const int l = (idx / Cout) % LLEN;
  const int b = idx / (Cout * LLEN);
  const int lm = (l + LLEN - 1) & (LLEN - 1);
  const int lp = (l + 1) & (LLEN - 1);
  const float* ib = in + (size_t)b * LLEN * Cin;
  const float* wb = w + (size_t)o * Cin * 3;
  float s = 0.f;
  for (int c = 0; c < Cin; ++c)
    s += ib[(size_t)lm * Cin + c] * wb[c * 3 + 0]
       + ib[(size_t)l  * Cin + c] * wb[c * 3 + 1]
       + ib[(size_t)lp * Cin + c] * wb[c * 3 + 2];
  if (addto) out[idx] += s; else out[idx] = s;
}

// moving mean (k=25, replicate pad) of [B,L,512] into mm
__global__ void k_movmean(const float* __restrict__ x, float* __restrict__ mm, int n)
{
  int idx = blockIdx.x * blockDim.x + threadIdx.x;
  if (idx >= n) return;
  const int d = idx % DMODEL;
  const int l = (idx / DMODEL) % LLEN;
  const int b = idx / (DMODEL * LLEN);
  const float* base = x + (size_t)b * LLEN * DMODEL + d;
  float s = 0.f;
#pragma unroll
  for (int j = -12; j <= 12; ++j) {
    int t = l + j; t = t < 0 ? 0 : (t > LLEN - 1 ? LLEN - 1 : t);
    s += base[(size_t)t * DMODEL];
  }
  mm[idx] = s * (1.0f / 25.0f);
}

__global__ void k_decomp_apply(float* __restrict__ x, const float* __restrict__ mm,
                               float* __restrict__ tsum, int n)
{
  int idx = blockIdx.x * blockDim.x + threadIdx.x;
  if (idx >= n) return;
  const float m = mm[idx];
  x[idx] -= m;
  if (tsum) tsum[idx] += m;
}

__global__ void k_gelu(float* __restrict__ h, int n)
{
  int idx = blockIdx.x * blockDim.x + threadIdx.x;
  if (idx >= n) return;
  const float v = h[idx];
  h[idx] = 0.5f * v * (1.0f + erff(v * 0.70710678118654752f));
}

__global__ void k_fill(float* __restrict__ p, float v, int n)
{
  int idx = blockIdx.x * blockDim.x + threadIdx.x;
  if (idx < n) p[idx] = v;
}

__global__ void k_ln_rows(float* __restrict__ x, const float* __restrict__ g,
                          const float* __restrict__ bb)
{
  int idx = blockIdx.x * blockDim.x + threadIdx.x;
  if (idx >= BBATCH * LLEN) return;
  float* p = x + (size_t)idx * DMODEL;
  float mu = 0.f;
  for (int d = 0; d < DMODEL; ++d) mu += p[d];
  mu *= (1.0f / DMODEL);
  float var = 0.f;
  for (int d = 0; d < DMODEL; ++d) { const float t = p[d] - mu; var += t * t; }
  var *= (1.0f / DMODEL);
  const float inv = rsqrtf(var + 1e-5f);
  for (int d = 0; d < DMODEL; ++d) p[d] = (p[d] - mu) * inv * g[d] + bb[d];
}

__global__ void k_colmean(const float* __restrict__ x, float* __restrict__ cm)
{
  int idx = blockIdx.x * blockDim.x + threadIdx.x;
  if (idx >= BBATCH * DMODEL) return;
  const int d = idx % DMODEL, b = idx / DMODEL;
  const float* base = x + (size_t)b * LLEN * DMODEL + d;
  float s = 0.f;
  for (int l = 0; l < LLEN; ++l) s += base[(size_t)l * DMODEL];
  cm[idx] = s * (1.0f / LLEN);
}

__global__ void k_subcm(float* __restrict__ x, const float* __restrict__ cm, int n)
{
  int idx = blockIdx.x * blockDim.x + threadIdx.x;
  if (idx >= n) return;
  const int d = idx % DMODEL;
  const int b = idx / (DMODEL * LLEN);
  x[idx] -= cm[b * DMODEL + d];
}

__global__ void k_meanL(const float* __restrict__ xe, float* __restrict__ meanb)
{
  int idx = blockIdx.x * blockDim.x + threadIdx.x;
  if (idx >= BBATCH * CCH) return;
  const int c = idx % CCH, b = idx / CCH;
  const float* base = xe + (size_t)b * LLEN * CCH + c;
  float s = 0.f;
  for (int l = 0; l < LLEN; ++l) s += base[(size_t)l * CCH];
  meanb[idx] = s * (1.0f / LLEN);
}

// seasonal_init / trend_init ([B,2048,21]): first 1024 from decomp tail of x_enc,
// last 1024: seasonal=0, trend=mean(x_enc).
__global__ void k_prep(const float* __restrict__ xe, const float* __restrict__ meanb,
                       float* __restrict__ sinit, float* __restrict__ trend)
{
  int idx = blockIdx.x * blockDim.x + threadIdx.x;
  if (idx >= BBATCH * LLEN * CCH) return;
  const int c = idx % CCH;
  const int l = (idx / CCH) % LLEN;
  const int b = idx / (CCH * LLEN);
  float se, tr;
  if (l < 1024) {
    const int src = 1024 + l;
    const float* base = xe + (size_t)b * LLEN * CCH + c;
    float s = 0.f;
    for (int j = -12; j <= 12; ++j) {
      int t = src + j; t = t < 0 ? 0 : (t > LLEN - 1 ? LLEN - 1 : t);
      s += base[(size_t)t * CCH];
    }
    const float mm = s * (1.0f / 25.0f);
    se = base[(size_t)src * CCH] - mm;
    tr = mm;
  } else {
    se = 0.f;
    tr = meanb[b * CCH + c];
  }
  sinit[idx] = se;
  trend[idx] = tr;
}

// out[b,t] = trend[b,1024+t,20] + x[b,1024+t,:] . proj_w[20,:] + proj_b[20]
__global__ void k_final(const float* __restrict__ x, const float* __restrict__ pw,
                        const float* __restrict__ pb, const float* __restrict__ trend,
                        float* __restrict__ out)
{
  int idx = blockIdx.x * blockDim.x + threadIdx.x;
  if (idx >= BBATCH * 1024) return;
  const int t = idx & 1023, b = idx >> 10;
  const int l = 1024 + t;
  const float* row  = x + ((size_t)b * LLEN + l) * DMODEL;
  const float* wrow = pw + 20 * DMODEL;
  float s = pb[20];
  for (int d = 0; d < DMODEL; ++d) s += row[d] * wrow[d];
  out[idx] = s + trend[((size_t)b * LLEN + l) * CCH + 20];
}

// ---------------------------------------------------------------------------
extern "C" void kernel_launch(void* const* d_in, const int* in_sizes, int n_in,
                              void* d_out, int out_size, void* d_ws, size_t ws_size,
                              hipStream_t stream)
{
  (void)in_sizes; (void)n_in; (void)out_size; (void)ws_size;
  const float* x_enc = (const float*)d_in[0];
  // params flattened in jax pytree order (sorted dict keys, lists in order):
  auto in = [&](int i) { return (const float*)d_in[i]; };
  const float* dec_emb_w  = in(4);
  const float* d0_c1      = in(5);
  const float* d0_c2      = in(6);
  const float* ca_bk      = in(7);
  const float* ca_bo      = in(8);
  const float* ca_bq      = in(9);
  const float* ca_wim     = in(10);
  const float* ca_wk      = in(11);
  const float* ca_wo      = in(12);
  const float* ca_wq      = in(13);
  const float* ca_wre     = in(14);
  const float* sa_bo      = in(15);
  const float* sa_bq      = in(16);
  const float* sa_wim     = in(17);
  const float* sa_wo      = in(18);
  const float* sa_wq      = in(19);
  const float* sa_wre     = in(20);
  const float* d0_trend_w = in(21);
  const float* dec_norm_b = in(22);
  const float* dec_norm_g = in(23);
  const float* dec_proj_b = in(24);
  const float* dec_proj_w = in(25);
  const float* enc_emb_w  = in(26);
  const float* enc_bo[2]  = { in(27), in(35) };
  const float* enc_bq[2]  = { in(28), in(36) };
  const float* enc_wim[2] = { in(29), in(37) };
  const float* enc_wo[2]  = { in(30), in(38) };
  const float* enc_wq[2]  = { in(31), in(39) };
  const float* enc_wre[2] = { in(32), in(40) };
  const float* enc_c1[2]  = { in(33), in(41) };
  const float* enc_c2[2]  = { in(34), in(42) };
  const float* enc_norm_b = in(43);
  const float* enc_norm_g = in(44);
  float* out = (float*)d_out;

  // workspace carve-up
  size_t off = 0;
  auto alloc = [&](size_t nf) {
    float* p = (float*)((char*)d_ws + off);
    off = (off + nf * sizeof(float) + 255) & ~(size_t)255;
    return p;
  };
  const size_t S = (size_t)BBATCH * LLEN * DMODEL;   // 16.78M floats
  float* xbuf   = alloc(S);          // decoder stream
  float* enc    = alloc(S);          // encoder stream / enc_out
  float* qt     = alloc(S);          // [8192,2048] transposed q; doubles as iDFT out
  float* hbuf   = alloc(S);          // q-proj out / FFN hidden chunk / movmean scratch
  float* tsum   = alloc(S);          // t1+t2+t3 (decoder)
  float* coeffA = alloc((size_t)NSEQ * 128);
  float* coeffB = alloc((size_t)NSEQ * 128);
  float* coeffC = alloc((size_t)NSEQ * 128);
  float* xqk    = alloc((size_t)BBATCH * NHEADS * 4096 * 2);
  float* fbasis = alloc((size_t)2 * NMODES * LLEN);
  float* ibasis = alloc((size_t)LLEN * 2 * NMODES);
  float* sinit  = alloc((size_t)BBATCH * LLEN * CCH);
  float* trendb = alloc((size_t)BBATCH * LLEN * CCH);
  float* colm   = alloc((size_t)BBATCH * DMODEL);
  float* meanb  = alloc((size_t)BBATCH * CCH);

  const int NROWS = BBATCH * LLEN;         // 32768
  const int NEL   = NROWS * DMODEL;        // 16.78M

  auto gemm = [&](const float* A, const float* W, const float* bias, float* C,
                  int M, int N, int K, int addto) {
    dim3 g(N / 64, M / 64), b(128);
    if (addto) {
      if (bias) k_gemm_bf16<true,  true ><<<g, b, 0, stream>>>(A, W, bias, C, M, N, K);
      else      k_gemm_bf16<true,  false><<<g, b, 0, stream>>>(A, W, bias, C, M, N, K);
    } else {
      if (bias) k_gemm_bf16<false, true ><<<g, b, 0, stream>>>(A, W, bias, C, M, N, K);
      else      k_gemm_bf16<false, false><<<g, b, 0, stream>>>(A, W, bias, C, M, N, K);
    }
  };
  auto transpose = [&](const float* inb, float* outb) {
    k_transpose_bl<<<dim3(LLEN / 32, DMODEL / 32, BBATCH), dim3(32, 8), 0, stream>>>(inb, outb);
  };
  auto L1D = [&](int n) { return dim3((n + 255) / 256); };

  auto self_block = [&](float* xb, const float* wq, const float* bq,
                        const float* wre, const float* wim,
                        const float* wo, const float* bo) {
    gemm(xb, wq, bq, hbuf, NROWS, DMODEL, DMODEL, 0);              // q = x@wq.T+bq
    transpose(hbuf, qt);                                           // [8192,2048]
    gemm(qt, fbasis, nullptr, coeffA, NSEQ, 2 * NMODES, LLEN, 0);  // DFT (64 modes)
    k_mode_mix<<<L1D(NSEQ * NMODES), 256, 0, stream>>>(coeffA, wre, wim, coeffB, 1.0f);
    gemm(coeffB, ibasis, nullptr, qt, NSEQ, LLEN, 2 * NMODES, 0);  // iDFT -> y (reinterp [32768,512])
    gemm(qt, wo, bo, xb, NROWS, DMODEL, DMODEL, 1);                // x += y@wo.T+bo
  };
  auto cross_block = [&](float* xb) {
    gemm(xb, ca_wq, ca_bq, hbuf, NROWS, DMODEL, DMODEL, 0);
    transpose(hbuf, qt);
    gemm(qt, fbasis, nullptr, coeffA, NSEQ, 2 * NMODES, LLEN, 0);  // qf
    gemm(enc, ca_wk, ca_bk, hbuf, NROWS, DMODEL, DMODEL, 0);
    transpose(hbuf, qt);
    gemm(qt, fbasis, nullptr, coeffB, NSEQ, 2 * NMODES, LLEN, 0);  // kf
    k_xqk_tanh<<<L1D(NSEQ * 64), 256, 0, stream>>>(coeffA, coeffB, xqk);
    k_xqkv<<<L1D(NSEQ * 64), 256, 0, stream>>>(xqk, coeffB, coeffC);
    k_mode_mix<<<L1D(NSEQ * NMODES), 256, 0, stream>>>(coeffC, ca_wre, ca_wim, coeffA,
                                                       1.0f / (512.0f * 512.0f));
    gemm(coeffA, ibasis, nullptr, qt, NSEQ, LLEN, 2 * NMODES, 0);
    gemm(qt, ca_wo, ca_bo, xb, NROWS, DMODEL, DMODEL, 1);
  };
  auto decomp = [&](float* xb, float* tacc) {
    k_movmean<<<L1D(NEL), 256, 0, stream>>>(xb, hbuf, NEL);
    k_decomp_apply<<<L1D(NEL), 256, 0, stream>>>(xb, hbuf, tacc, NEL);
  };
  auto ffn = [&](float* xb, const float* c1, const float* c2) {
    for (int ch = 0; ch < 4; ++ch) {                 // chunk rows: hidden fits hbuf
      float* xrow = xb + (size_t)ch * 8192 * DMODEL;
      gemm(xrow, c1, nullptr, hbuf, 8192, DFF, DMODEL, 0);
      k_gelu<<<L1D(8192 * DFF), 256, 0, stream>>>(hbuf, 8192 * DFF);
      gemm(hbuf, c2, nullptr, xrow, 8192, DMODEL, DFF, 1);
    }
  };
  auto mylayernorm = [&](float* xb, const float* g, const float* b) {
    k_ln_rows<<<L1D(NROWS), 256, 0, stream>>>(xb, g, b);
    k_colmean<<<L1D(BBATCH * DMODEL), 256, 0, stream>>>(xb, colm);
    k_subcm<<<L1D(NEL), 256, 0, stream>>>(xb, colm, NEL);
  };

  // ---- init: bases, decomposition of x_enc ----
  k_gen_basis<<<L1D(NMODES * LLEN), 256, 0, stream>>>(fbasis, ibasis);
  k_meanL<<<L1D(BBATCH * CCH), 256, 0, stream>>>(x_enc, meanb);
  k_prep<<<L1D(BBATCH * LLEN * CCH), 256, 0, stream>>>(x_enc, meanb, sinit, trendb);

  // ---- encoder ----
  k_conv3<<<L1D(NROWS * DMODEL), 256, 0, stream>>>(x_enc, enc_emb_w, enc, CCH, DMODEL, 0);
  for (int layer = 0; layer < 2; ++layer) {
    self_block(enc, enc_wq[layer], enc_bq[layer], enc_wre[layer], enc_wim[layer],
               enc_wo[layer], enc_bo[layer]);
    decomp(enc, nullptr);
    ffn(enc, enc_c1[layer], enc_c2[layer]);
    decomp(enc, nullptr);
  }
  mylayernorm(enc, enc_norm_g, enc_norm_b);

  // ---- decoder (1 layer) ----
  k_conv3<<<L1D(NROWS * DMODEL), 256, 0, stream>>>(sinit, dec_emb_w, xbuf, CCH, DMODEL, 0);
  k_fill<<<L1D(NEL), 256, 0, stream>>>(tsum, 0.0f, NEL);
  self_block(xbuf, sa_wq, sa_bq, sa_wre, sa_wim, sa_wo, sa_bo);
  decomp(xbuf, tsum);
  cross_block(xbuf);
  decomp(xbuf, tsum);
  ffn(xbuf, d0_c1, d0_c2);
  decomp(xbuf, tsum);
  k_conv3<<<L1D(NROWS * CCH), 256, 0, stream>>>(tsum, d0_trend_w, trendb, DMODEL, CCH, 1);
  mylayernorm(xbuf, dec_norm_g, dec_norm_b);
  k_final<<<L1D(BBATCH * 1024), 256, 0, stream>>>(xbuf, dec_proj_w, dec_proj_b, trendb, out);
}
